// Spectral_AttnBlock_14190571946266
// MI455X (gfx1250) — matrix operand order repaired and checked by reading, not compiled, not run on hardware
//
#include <hip/hip_runtime.h>
#include <hip/hip_bf16.h>
#include <math.h>

// ---------------------------------------------------------------------------
// Problem constants
// ---------------------------------------------------------------------------
#define BB   8
#define CC   256
#define HH   160
#define WWID 160
#define SS   (HH * WWID)          // 25600
#define HEADS 32
#define DHEAD 8
#define GROUPS 16

typedef __attribute__((ext_vector_type(16))) __bf16 v16bf;
typedef __attribute__((ext_vector_type(8)))  float  v8f;

// ---------------------------------------------------------------------------
// bf16 helpers (bit-level, round-to-nearest-even)
// ---------------------------------------------------------------------------
__device__ __forceinline__ __bf16 f2bf(float f) {
  unsigned u = __float_as_uint(f);
  unsigned r = (u + 0x7FFFu + ((u >> 16) & 1u)) >> 16;
  unsigned short h = (unsigned short)r;
  __bf16 out;
  __builtin_memcpy(&out, &h, 2);
  return out;
}
__device__ __forceinline__ float bf2f(__bf16 b) {
  unsigned short h;
  __builtin_memcpy(&h, &b, 2);
  return __uint_as_float(((unsigned)h) << 16);
}

__device__ __forceinline__ float wave_sum(float v) {
#pragma unroll
  for (int off = 16; off > 0; off >>= 1) v += __shfl_xor(v, off, 32);
  return v;
}

// ---------------------------------------------------------------------------
// WMMA helpers (CDNA5 bf16 16x16x32, wave32)
// A (16x32, MxK): lane m = l&15, half = l>>4.
//   element j: k = half*16 + j
// B (32x16, KxN): lane n = l&15, half = l>>4; element j: k = half*16 + j
// C/D (16x16 f32): VGPR i, lane l: M = i + 8*(l>>4), N = l&15.
// ---------------------------------------------------------------------------
__device__ __forceinline__ v16bf load_a_frag(const __bf16* row, int k0, int half) {
  v16bf a;
  ((uint4*)&a)[0] = *(const uint4*)(row + k0 + half * 16);
  ((uint4*)&a)[1] = *(const uint4*)(row + k0 + half * 16 + 8);
  return a;
}
__device__ __forceinline__ v16bf load_b_frag(const __bf16* wrow, int k0, int half) {
  v16bf b;
  const uint4* p = (const uint4*)(wrow + k0 + half * 16);
  ((uint4*)&b)[0] = p[0];
  ((uint4*)&b)[1] = p[1];
  return b;
}
__device__ __forceinline__ v8f wmma_bf16(v16bf a, v16bf b, v8f c) {
  return __builtin_amdgcn_wmma_f32_16x16x32_bf16(false, a, false, b, (short)0, c,
                                                 false, false);
}

// ---------------------------------------------------------------------------
// 1) GroupNorm statistics: one block per (group, batch)
// ---------------------------------------------------------------------------
__global__ void gn_stats_k(const float* __restrict__ x, float* __restrict__ stats) {
  const int g = blockIdx.x, b = blockIdx.y;
  const float* base = x + ((size_t)b * CC + (size_t)g * 16) * SS;
  float sum = 0.f, ssq = 0.f;
  for (int ci = 0; ci < 16; ++ci) {
    const float* row = base + (size_t)ci * SS;
    for (int s = threadIdx.x; s < SS; s += 256) {
      float v = row[s];
      sum += v;
      ssq += v * v;
    }
  }
  __shared__ float r0[8], r1[8];
  float ws0 = wave_sum(sum), ws1 = wave_sum(ssq);
  int wave = threadIdx.x >> 5, lane = threadIdx.x & 31;
  if (lane == 0) { r0[wave] = ws0; r1[wave] = ws1; }
  __syncthreads();
  if (threadIdx.x == 0) {
    float s0 = 0.f, s1 = 0.f;
    for (int i = 0; i < 8; ++i) { s0 += r0[i]; s1 += r1[i]; }
    const float inv_n = 1.0f / (16.0f * SS);
    float mean = s0 * inv_n;
    float var = s1 * inv_n - mean * mean;
    stats[(b * GROUPS + g) * 2 + 0] = mean;
    stats[(b * GROUPS + g) * 2 + 1] = rsqrtf(var + 1e-5f);
  }
}

// ---------------------------------------------------------------------------
// 2) Apply GN + pack bf16, transpose [b,c,s] -> [b,s,c]
// ---------------------------------------------------------------------------
__global__ void gn_apply_k(const float* __restrict__ x, const float* __restrict__ stats,
                           const float* __restrict__ gw, const float* __restrict__ gb,
                           __bf16* __restrict__ xf) {
  const int b = blockIdx.z;
  const int c = blockIdx.y * 32 + (threadIdx.x & 31);
  const int s0 = blockIdx.x * 32;
  const int ty = threadIdx.x >> 5;  // 0..7
  const int gi = c >> 4;
  const float mean = stats[(b * GROUPS + gi) * 2 + 0];
  const float rstd = stats[(b * GROUPS + gi) * 2 + 1];
  const float sc = gw[c] * rstd;
  const float bi = gb[c] - mean * sc;
  const float* xrow = x + ((size_t)b * CC + c) * SS;
#pragma unroll
  for (int it = 0; it < 4; ++it) {
    int s = s0 + ty + it * 8;
    float v = xrow[s];
    xf[((size_t)b * SS + s) * CC + c] = f2bf(v * sc + bi);
  }
}

// ---------------------------------------------------------------------------
// 3) Convert Wq/Wk/Wv f32 -> bf16
// ---------------------------------------------------------------------------
__global__ void wcvt_k(const float* __restrict__ wq, const float* __restrict__ wk,
                       const float* __restrict__ wv, __bf16* __restrict__ o) {
  int i = blockIdx.x * 256 + threadIdx.x;  // 65536 total
  o[i] = f2bf(wq[i]);
  o[65536 + i] = f2bf(wk[i]);
  o[131072 + i] = f2bf(wv[i]);
}

// ---------------------------------------------------------------------------
// 4) QKV GEMM (WMMA bf16): three accumulator chains interleaved per k-step
//    so loads overlap wmma. q,k -> [b,c,s] bf16; v -> [b,s,c] bf16.
//    grid (S/16/8, B), block 256 (8 waves, one m-tile per wave)
// ---------------------------------------------------------------------------
__global__ void qkv_gemm_k(const __bf16* __restrict__ xf, const __bf16* __restrict__ wb,
                           __bf16* __restrict__ qT, __bf16* __restrict__ kT,
                           __bf16* __restrict__ vO) {
  const int wave = threadIdx.x >> 5;
  const int lane = threadIdx.x & 31;
  const int mtile = blockIdx.x * 8 + wave;  // 0..1599
  const int b = blockIdx.y;
  const int m = lane & 15, half = lane >> 4;
  const int n = lane & 15;

  const __bf16* arow = xf + ((size_t)b * SS + (size_t)mtile * 16 + m) * CC;
  v16bf afrag[8];
#pragma unroll
  for (int ks = 0; ks < 8; ++ks) afrag[ks] = load_a_frag(arow, ks * 32, half);

  const __bf16* Wq = wb;
  const __bf16* Wk = wb + 65536;
  const __bf16* Wv = wb + 131072;

  for (int nt = 0; nt < 16; ++nt) {
    const int c = nt * 16 + n;
    const __bf16* wqrow = Wq + (size_t)c * CC;
    const __bf16* wkrow = Wk + (size_t)c * CC;
    const __bf16* wvrow = Wv + (size_t)c * CC;
    v8f aq = {}, ak = {}, av = {};
#pragma unroll
    for (int ks = 0; ks < 8; ++ks) {
      v16bf bq = load_b_frag(wqrow, ks * 32, half);
      v16bf bk = load_b_frag(wkrow, ks * 32, half);
      v16bf bv = load_b_frag(wvrow, ks * 32, half);
      aq = wmma_bf16(afrag[ks], bq, aq);
      ak = wmma_bf16(afrag[ks], bk, ak);
      av = wmma_bf16(afrag[ks], bv, av);
    }
    {
      __bf16 tmp[8];
#pragma unroll
      for (int i = 0; i < 8; ++i) tmp[i] = f2bf(aq[i]);
      *(uint4*)(qT + ((size_t)b * CC + c) * SS + (size_t)mtile * 16 + half * 8) =
          *(const uint4*)tmp;
    }
    {
      __bf16 tmp[8];
#pragma unroll
      for (int i = 0; i < 8; ++i) tmp[i] = f2bf(ak[i]);
      *(uint4*)(kT + ((size_t)b * CC + c) * SS + (size_t)mtile * 16 + half * 8) =
          *(const uint4*)tmp;
    }
#pragma unroll
    for (int i = 0; i < 8; ++i) {
      int s = mtile * 16 + half * 8 + i;
      vO[((size_t)b * SS + s) * CC + c] = f2bf(av[i]);
    }
  }
}

// ---------------------------------------------------------------------------
// 5) Gram matrices + squared norms: one block per (b, head)
//    per bh: [0..63]=G[i*8+j], [64..71]=sum k_i^2, [72..79]=sum q_j^2
// ---------------------------------------------------------------------------
__global__ void gram_k(const __bf16* __restrict__ qT, const __bf16* __restrict__ kT,
                       float* __restrict__ G) {
  const int bh = blockIdx.x;
  const int b = bh >> 5, h = bh & 31;
  const __bf16* qbase = qT + ((size_t)b * CC + (size_t)h * 8) * SS;
  const __bf16* kbase = kT + ((size_t)b * CC + (size_t)h * 8) * SS;

  float acc[8][8] = {};
  float kss[8] = {}, qss[8] = {};
  for (int s = threadIdx.x; s < SS; s += 256) {
    float qf[8], kf[8];
#pragma unroll
    for (int i = 0; i < 8; ++i) {
      qf[i] = bf2f(qbase[(size_t)i * SS + s]);
      kf[i] = bf2f(kbase[(size_t)i * SS + s]);
    }
#pragma unroll
    for (int i = 0; i < 8; ++i) {
#pragma unroll
      for (int j = 0; j < 8; ++j) acc[i][j] = fmaf(kf[i], qf[j], acc[i][j]);
      kss[i] = fmaf(kf[i], kf[i], kss[i]);
      qss[i] = fmaf(qf[i], qf[i], qss[i]);
    }
  }
  __shared__ float red[80];
  if (threadIdx.x < 80) red[threadIdx.x] = 0.f;
  __syncthreads();
  const int lane = threadIdx.x & 31;
#pragma unroll
  for (int i = 0; i < 8; ++i)
#pragma unroll
    for (int j = 0; j < 8; ++j) {
      float r = wave_sum(acc[i][j]);
      if (lane == 0) atomicAdd(&red[i * 8 + j], r);
    }
#pragma unroll
  for (int i = 0; i < 8; ++i) {
    float rk = wave_sum(kss[i]);
    if (lane == 0) atomicAdd(&red[64 + i], rk);
    float rq = wave_sum(qss[i]);
    if (lane == 0) atomicAdd(&red[72 + i], rq);
  }
  __syncthreads();
  if (threadIdx.x < 80) G[(size_t)bh * 80 + threadIdx.x] = red[threadIdx.x];
}

// ---------------------------------------------------------------------------
// 6) attn = softmax_j( G_ij / (||k_i|| ||q_j||) * rescale[h] )
// ---------------------------------------------------------------------------
__global__ void attn_k(const float* __restrict__ G, const float* __restrict__ rescale,
                       float* __restrict__ attn) {
  const int bh = blockIdx.x;
  const int h = bh & 31;
  const float* g = G + (size_t)bh * 80;
  __shared__ float pre[64], rmax[8], rsum[8];
  const int t = threadIdx.x;  // 64 threads
  const int i = t >> 3, j = t & 7;
  float nk = sqrtf(g[64 + i]); nk = fmaxf(nk, 1e-12f);
  float nq = sqrtf(g[72 + j]); nq = fmaxf(nq, 1e-12f);
  pre[t] = g[t] / (nk * nq) * rescale[h];
  __syncthreads();
  if (t < 8) {
    float m = -3.4e38f;
    for (int jj = 0; jj < 8; ++jj) m = fmaxf(m, pre[t * 8 + jj]);
    float s = 0.f;
    for (int jj = 0; jj < 8; ++jj) s += expf(pre[t * 8 + jj] - m);
    rmax[t] = m;
    rsum[t] = s;
  }
  __syncthreads();
  attn[(size_t)bh * 64 + t] = expf(pre[t] - rmax[i]) / rsum[i];
}

// ---------------------------------------------------------------------------
// 7) Compose M_b = proj_w @ blockdiag(attn_b)  -> bf16 [b][o][c]
// ---------------------------------------------------------------------------
__global__ void composeM_k(const float* __restrict__ attn, const float* __restrict__ proj_w,
                           __bf16* __restrict__ Mmat) {
  const int o = blockIdx.x, b = blockIdx.y;
  const int cj = threadIdx.x;
  const int h = cj >> 3, j = cj & 7;
  const float* pw = proj_w + (size_t)o * CC + h * 8;
  const float* at = attn + ((size_t)(b * 32 + h)) * 64 + j;
  float acc = 0.f;
#pragma unroll
  for (int i = 0; i < 8; ++i) acc = fmaf(pw[i], at[i * 8], acc);
  Mmat[((size_t)b * CC + o) * CC + cj] = f2bf(acc);
}

// ---------------------------------------------------------------------------
// 8) Depthwise conv 3x3 on v ([b,s,c] bf16) + exact GELU -> g [b,c,s] bf16
//    LDS-tiled: block = (row y, 32 channels, batch). Stages 3 rows x 160 x 32c
//    of v (coalesced), computes with lanes along x so the [b,c,s] stores are
//    contiguous 64B per wave. LDS inner stride 34 (17-bank lane stride,
//    conflict-free).
// ---------------------------------------------------------------------------
__global__ void dwconv1_k(const __bf16* __restrict__ v, const float* __restrict__ w1,
                          __bf16* __restrict__ g) {
  const int y = blockIdx.x;
  const int c0 = blockIdx.y * 32;
  const int b = blockIdx.z;
  __shared__ __bf16 lv[3 * 160 * 34];

  for (int e = threadIdx.x; e < 3 * 160 * 32; e += 256) {
    int cc = e & 31;
    int pos = e >> 5;        // 0..479
    int xx = pos % 160;
    int ry = pos / 160;      // 0..2
    int yy = y + ry - 1;
    __bf16 val = f2bf(0.f);
    if (yy >= 0 && yy < HH)
      val = v[((size_t)b * SS + (size_t)yy * WWID + xx) * CC + c0 + cc];
    lv[(ry * 160 + xx) * 34 + cc] = val;
  }
  __syncthreads();

  const int lane = threadIdx.x & 31;
  const int cgrp = (threadIdx.x >> 5) * 4;  // wave -> 4 channels
#pragma unroll
  for (int ci = 0; ci < 4; ++ci) {
    const int cc = cgrp + ci;
    const int c = c0 + cc;
    float w[9];
#pragma unroll
    for (int t = 0; t < 9; ++t) w[t] = w1[c * 9 + t];
#pragma unroll
    for (int xi = 0; xi < 5; ++xi) {
      const int x = lane + xi * 32;
      float acc = 0.f;
#pragma unroll
      for (int ry = 0; ry < 3; ++ry) {
#pragma unroll
        for (int dx = -1; dx <= 1; ++dx) {
          int xx = x + dx;
          if (xx < 0 || xx >= WWID) continue;
          acc = fmaf(w[ry * 3 + dx + 1], bf2f(lv[(ry * 160 + xx) * 34 + cc]), acc);
        }
      }
      float ge = 0.5f * acc * (1.0f + erff(acc * 0.70710678118654752f));
      g[((size_t)b * CC + c) * SS + (size_t)y * WWID + x] = f2bf(ge);
    }
  }
}

// ---------------------------------------------------------------------------
// 9) out GEMM (WMMA bf16): out_c = v @ M_b^T + proj_b  -> d_out f32 [b,c,s]
//    two n-tiles in flight per iteration (two independent acc chains)
// ---------------------------------------------------------------------------
__global__ void out_gemm_k(const __bf16* __restrict__ v, const __bf16* __restrict__ Mmat,
                           const float* __restrict__ proj_b, float* __restrict__ out) {
  const int wave = threadIdx.x >> 5;
  const int lane = threadIdx.x & 31;
  const int mtile = blockIdx.x * 8 + wave;
  const int b = blockIdx.y;
  const int m = lane & 15, half = lane >> 4;
  const int n = lane & 15;

  const __bf16* arow = v + ((size_t)b * SS + (size_t)mtile * 16 + m) * CC;
  v16bf afrag[8];
#pragma unroll
  for (int ks = 0; ks < 8; ++ks) afrag[ks] = load_a_frag(arow, ks * 32, half);

  const __bf16* Mb = Mmat + (size_t)b * CC * CC;
  for (int nt = 0; nt < 16; nt += 2) {
    const int ca = nt * 16 + n;
    const int cb = (nt + 1) * 16 + n;
    const __bf16* wra = Mb + (size_t)ca * CC;
    const __bf16* wrb = Mb + (size_t)cb * CC;
    v8f acc0 = {}, acc1 = {};
#pragma unroll
    for (int ks = 0; ks < 8; ++ks) {
      v16bf b0 = load_b_frag(wra, ks * 32, half);
      v16bf b1 = load_b_frag(wrb, ks * 32, half);
      acc0 = wmma_bf16(afrag[ks], b0, acc0);
      acc1 = wmma_bf16(afrag[ks], b1, acc1);
    }
    {
      const float bias = proj_b[ca];
      float* op = out + ((size_t)b * CC + ca) * SS + (size_t)mtile * 16 + half * 8;
      *(float4*)op = make_float4(acc0[0] + bias, acc0[1] + bias, acc0[2] + bias, acc0[3] + bias);
      *(float4*)(op + 4) = make_float4(acc0[4] + bias, acc0[5] + bias, acc0[6] + bias, acc0[7] + bias);
    }
    {
      const float bias = proj_b[cb];
      float* op = out + ((size_t)b * CC + cb) * SS + (size_t)mtile * 16 + half * 8;
      *(float4*)op = make_float4(acc1[0] + bias, acc1[1] + bias, acc1[2] + bias, acc1[3] + bias);
      *(float4*)(op + 4) = make_float4(acc1[4] + bias, acc1[5] + bias, acc1[6] + bias, acc1[7] + bias);
    }
  }
}

// ---------------------------------------------------------------------------
// 10) out += dwconv2(g) + x  (everything in [b,c,s] layout, coalesced)
// ---------------------------------------------------------------------------
__global__ void dwconv2_add_k(const __bf16* __restrict__ g, const float* __restrict__ w2,
                              const float* __restrict__ x, float* __restrict__ out) {
  const size_t idx = (size_t)blockIdx.x * 256 + threadIdx.x;  // b*c*s linear
  const int s = (int)(idx % SS);
  const size_t bc = idx / SS;
  const int c = (int)(bc % CC);
  const int y = s / WWID, xx0 = s % WWID;
  const __bf16* gp = g + bc * SS;
  float acc = 0.f;
#pragma unroll
  for (int dy = -1; dy <= 1; ++dy) {
    int yy = y + dy;
    if (yy < 0 || yy >= HH) continue;
#pragma unroll
    for (int dx = -1; dx <= 1; ++dx) {
      int xx = xx0 + dx;
      if (xx < 0 || xx >= WWID) continue;
      acc = fmaf(w2[c * 9 + (dy + 1) * 3 + (dx + 1)], bf2f(gp[yy * WWID + xx]), acc);
    }
  }
  out[idx] = out[idx] + acc + x[idx];
}

// ---------------------------------------------------------------------------
// Workspace layout (bytes, 256-aligned)
// ---------------------------------------------------------------------------
#define OFF_STATS 0ull                      // 256 f32
#define OFF_GRAM  1024ull                   // 256*80 f32
#define OFF_ATTN  82944ull                  // 256*64 f32
#define OFF_WB    148480ull                 // 3*65536 bf16
#define OFF_MMAT  541696ull                 // 8*65536 bf16
#define OFF_XF    1590272ull                // B*S*C bf16 (reused as g)
#define OFF_QT    (OFF_XF + 104857600ull)
#define OFF_KT    (OFF_QT + 104857600ull)
#define OFF_V     (OFF_KT + 104857600ull)

extern "C" void kernel_launch(void* const* d_in, const int* in_sizes, int n_in,
                              void* d_out, int out_size, void* d_ws, size_t ws_size,
                              hipStream_t stream) {
  const float* x       = (const float*)d_in[0];
  const float* gn_w    = (const float*)d_in[1];
  const float* gn_b    = (const float*)d_in[2];
  const float* Wq      = (const float*)d_in[3];
  const float* Wk      = (const float*)d_in[4];
  const float* Wv      = (const float*)d_in[5];
  const float* rescale = (const float*)d_in[6];
  const float* proj_w  = (const float*)d_in[7];
  const float* proj_b  = (const float*)d_in[8];
  const float* conv1_w = (const float*)d_in[9];
  const float* conv2_w = (const float*)d_in[10];
  float* out = (float*)d_out;

  char* ws = (char*)d_ws;
  float*  stats = (float*)(ws + OFF_STATS);
  float*  gram  = (float*)(ws + OFF_GRAM);
  float*  attn  = (float*)(ws + OFF_ATTN);
  __bf16* wb    = (__bf16*)(ws + OFF_WB);
  __bf16* Mmat  = (__bf16*)(ws + OFF_MMAT);
  __bf16* xf    = (__bf16*)(ws + OFF_XF);
  __bf16* gbuf  = xf;  // xf is dead after qkv_gemm; reuse as dwconv1 output
  __bf16* qT    = (__bf16*)(ws + OFF_QT);
  __bf16* kT    = (__bf16*)(ws + OFF_KT);
  __bf16* vbuf  = (__bf16*)(ws + OFF_V);

  gn_stats_k   <<<dim3(GROUPS, BB),         256, 0, stream>>>(x, stats);
  gn_apply_k   <<<dim3(SS / 32, 8, BB),     256, 0, stream>>>(x, stats, gn_w, gn_b, xf);
  wcvt_k       <<<dim3(256),                256, 0, stream>>>(Wq, Wk, Wv, wb);
  qkv_gemm_k   <<<dim3(SS / 16 / 8, BB),    256, 0, stream>>>(xf, wb, qT, kT, vbuf);
  gram_k       <<<dim3(BB * HEADS),         256, 0, stream>>>(qT, kT, gram);
  attn_k       <<<dim3(BB * HEADS),          64, 0, stream>>>(gram, rescale, attn);
  composeM_k   <<<dim3(CC, BB),             256, 0, stream>>>(attn, proj_w, Mmat);
  dwconv1_k    <<<dim3(HH, CC / 32, BB),    256, 0, stream>>>(vbuf, conv1_w, gbuf);
  out_gemm_k   <<<dim3(SS / 16 / 8, BB),    256, 0, stream>>>(vbuf, Mmat, proj_b, out);
  dwconv2_add_k<<<dim3(BB * CC * SS / 256), 256, 0, stream>>>(gbuf, conv2_w, x, out);
}
